// DCNSepHead_42898133353397
// MI455X (gfx1250) — matrix operand order repaired and checked by reading, not compiled, not run on hardware
//
#include <hip/hip_runtime.h>
#include <hip/hip_bf16.h>

typedef _Float16 half16 __attribute__((ext_vector_type(16)));
typedef float    v8f    __attribute__((ext_vector_type(8)));

#define Hc 160
#define Wc 160
#define Bc 4
#define Cc 64
#define NPIX (Bc*Hc*Wc)        /* 102400 */
#define OUTC 13

// ---------------------------------------------------------------------------
// Pack fp32 weights [O][K] row-major into WMMA B-fragment order (f16).
// Fragment (kk, t): 32 lanes x 16 halves. Lane l covers N = 16*t + (l&15),
// K = 32*kk + (l>>4)*16 + h  (contiguous k order used here: lanes 0-15 hold K 0-15,
// lanes 16-31 hold K 16-31, VGPR v = K pair 2v,2v+1).
// ---------------------------------------------------------------------------
__global__ void pack_wB(const float* __restrict__ w, _Float16* __restrict__ pk,
                        int O, int K, int numNT) {
  int total = (K >> 5) * numNT * 512;
  int idx = blockIdx.x * blockDim.x + threadIdx.x;
  if (idx >= total) return;
  int kk = idx / (numNT * 512);
  int r  = idx % (numNT * 512);
  int t  = r / 512;
  int l  = (r % 512) >> 4;
  int h  = r & 15;
  int o  = 16 * t + (l & 15);
  int k  = 32 * kk + ((l >> 4) << 4) + h;
  float v = (o < O) ? w[o * K + k] : 0.f;
  pk[idx] = (_Float16)v;
}

// ---------------------------------------------------------------------------
// 1x1 offset conv (fp32, precision-sensitive: drives bilinear positions)
// ---------------------------------------------------------------------------
__global__ void offset_conv(const float* __restrict__ x, const float* __restrict__ w,
                            const float* __restrict__ bias, float* __restrict__ off) {
  long t = (long)blockIdx.x * blockDim.x + threadIdx.x;
  if (t >= (long)NPIX * 72) return;
  int pix = (int)(t / 72), ch = (int)(t % 72);
  int b = pix / (Hc * Wc), hw = pix % (Hc * Wc);
  const float* xp = x + (long)b * Cc * Hc * Wc + hw;
  const float* wp = w + ch * Cc;
  float acc = bias[ch];
#pragma unroll 8
  for (int c = 0; c < Cc; ++c) acc += xp[(long)c * Hc * Wc] * wp[c];
  off[(long)pix * 72 + ch] = acc;
}

// ---------------------------------------------------------------------------
// A-tile K-slice fill: 64 pixels x 64 K values into LDS (fp16).
// mode: 0 = deform bilinear gather, 1 = 3x3 im2col (fp16 NHWC), 2 = row copy.
// ---------------------------------------------------------------------------
__device__ __forceinline__ void fill_slice(_Float16* __restrict__ dst, int p0, int s,
                                           int mode, const float* __restrict__ x,
                                           const float* __restrict__ offs,
                                           const _Float16* __restrict__ featIn, int tid) {
  const int kbase = s * 64;
  for (int idx = tid; idx < 64 * 64; idx += 128) {
    int pp = idx >> 6, kl = idx & 63;
    int k = kbase + kl;
    int pix = p0 + pp;
    int b  = pix / (Hc * Wc), hw = pix % (Hc * Wc);
    int yq = hw / Wc, xq = hw % Wc;
    float v = 0.f;
    if (mode == 0) {
      int g = k / 144, cg = (k / 9) & 15, tap = k % 9;
      int c = g * 16 + cg;
      int ky = tap / 3, kx = tap % 3;
      const float* op = offs + (long)pix * 72 + (g * 9 + tap) * 2;
      float py = (float)(yq + ky - 1) + op[0];
      float px = (float)(xq + kx - 1) + op[1];
      float fy = floorf(py), fx = floorf(px);
      int y0 = (int)fy, x0 = (int)fx;
      float wy = py - fy, wx = px - fx;
      const float* xb = x + ((long)b * Cc + c) * (Hc * Wc);
#pragma unroll
      for (int dyc = 0; dyc < 2; ++dyc)
#pragma unroll
        for (int dxc = 0; dxc < 2; ++dxc) {
          int yy = y0 + dyc, xx = x0 + dxc;
          if (yy >= 0 && yy < Hc && xx >= 0 && xx < Wc) {
            float wgt = (dyc ? wy : 1.f - wy) * (dxc ? wx : 1.f - wx);
            v += xb[yy * Wc + xx] * wgt;
          }
        }
    } else if (mode == 1) {
      int c = k / 9, tap = k % 9;
      int yy = yq + tap / 3 - 1, xx = xq + tap % 3 - 1;
      if (yy >= 0 && yy < Hc && xx >= 0 && xx < Wc)
        v = (float)featIn[((long)b * Hc * Wc + yy * Wc + xx) * 64 + c];
    } else {
      v = (float)featIn[(long)pix * 64 + k];
    }
    dst[pp * 64 + kl] = (_Float16)v;
  }
}

// ---------------------------------------------------------------------------
// Implicit-GEMM, WMMA f16->f32. Block = 64 pixels (4 M-subtiles), 128 thr.
// A staged via double-buffered 64x64 fp16 LDS K-slices (16 KB, 1 barrier per
// slice). numNT==4: wave owns one N-tile, 4 accumulators over M-subtiles
// (each B fragment amortized over 4 WMMAs). numNT==1: wave owns one M-subtile.
// epi: 0=(+bias)ReLU->f16 NHWC  1=(+bias)BN ReLU->f16  2=(+bias)->f32 NCHW out
// ---------------------------------------------------------------------------
__launch_bounds__(128)
__global__ void wmma_gemm(int mode, const float* __restrict__ x,
                          const float* __restrict__ offs,
                          const _Float16* __restrict__ featIn,
                          const _Float16* __restrict__ packW,
                          int K, int numNT, int epi,
                          _Float16* __restrict__ featOut,
                          float* __restrict__ outF, int chOff, int nReal,
                          const float* __restrict__ bias,
                          const float* __restrict__ bng, const float* __restrict__ bnb,
                          const float* __restrict__ bnm, const float* __restrict__ bnv) {
  __shared__ _Float16 lA[2][64 * 64];     // double-buffered A K-slice (16 KB)
  const int p0   = blockIdx.x * 64;
  const int tid  = threadIdx.x;
  const int wave = tid >> 5, lane = tid & 31;
  const int hi   = lane >> 4;
  const int S    = K >> 6;                // 64-wide K slices (9 or 1)
  const int nkkg = K >> 5;

  v8f acc[4];
#pragma unroll
  for (int i = 0; i < 4; ++i) acc[i] = (v8f){0.f, 0.f, 0.f, 0.f, 0.f, 0.f, 0.f, 0.f};

  fill_slice(&lA[0][0], p0, 0, mode, x, offs, featIn, tid);
  __syncthreads();

  for (int s = 0; s < S; ++s) {
    const _Float16* buf = &lA[s & 1][0];
    if (numNT == 4) {
#pragma unroll
      for (int kk = 0; kk < 2; ++kk) {
        int kkg = s * 2 + kk;
        half16 bfr = *(const half16*)(packW + ((long)(kkg * 4 + wave) * 512 + lane * 16));
        if (kkg + 1 < nkkg)
          __builtin_prefetch(packW + ((long)((kkg + 1) * 4 + wave) * 512 + lane * 16), 0, 1);
#pragma unroll
        for (int ms = 0; ms < 4; ++ms) {
          half16 a;
          const _Float16* arow = buf + (ms * 16 + (lane & 15)) * 64 + kk * 32;
#pragma unroll
          for (int vv = 0; vv < 8; ++vv) {
            int k0 = hi * 16 + 2 * vv;
            a[2 * vv]     = arow[k0];
            a[2 * vv + 1] = arow[k0 + 1];
          }
          acc[ms] = __builtin_amdgcn_wmma_f32_16x16x32_f16(
              false, a, false, bfr, (short)0, acc[ms], false, false);
        }
      }
    } else {
#pragma unroll
      for (int kk = 0; kk < 2; ++kk) {
        int kkg = s * 2 + kk;
        half16 bfr = *(const half16*)(packW + ((long)kkg * 512 + lane * 16));
        if (kkg + 1 < nkkg)
          __builtin_prefetch(packW + ((long)(kkg + 1) * 512 + lane * 16), 0, 1);
        half16 a;
        const _Float16* arow = buf + (wave * 16 + (lane & 15)) * 64 + kk * 32;
#pragma unroll
        for (int vv = 0; vv < 8; ++vv) {
          int k0 = hi * 16 + 2 * vv;
          a[2 * vv]     = arow[k0];
          a[2 * vv + 1] = arow[k0 + 1];
        }
        acc[0] = __builtin_amdgcn_wmma_f32_16x16x32_f16(
            false, a, false, bfr, (short)0, acc[0], false, false);
      }
    }
    if (s + 1 < S) fill_slice(&lA[(s + 1) & 1][0], p0, s + 1, mode, x, offs, featIn, tid);
    __syncthreads();
  }

  // ---- epilogue: lane holds N = 16t+(l&15), M = vv + 8*hi within subtile ----
  const int nOwn = (numNT == 4) ? 4 : 1;
  for (int i = 0; i < nOwn; ++i) {
    int ms = (numNT == 4) ? i : wave;
    int t  = (numNT == 4) ? wave : 0;
    int n  = 16 * t + (lane & 15);
    float bvn = bias ? bias[n] : 0.f;
    if (epi == 2) {
      if ((lane & 15) < nReal) {
#pragma unroll
        for (int vv = 0; vv < 8; ++vv) {
          int pix = p0 + ms * 16 + vv + hi * 8;
          int b = pix / (Hc * Wc), hw = pix % (Hc * Wc);
          outF[((long)(b * OUTC + chOff + n) * (Hc * Wc)) + hw] = acc[i][vv] + bvn;
        }
      }
    } else {
      float scale = 1.f, shift = bvn;
      if (epi == 1) {
        float inv = rsqrtf(bnv[n] + 1e-5f);
        scale = bng[n] * inv;
        shift = scale * bvn + (bnb[n] - bng[n] * bnm[n] * inv);
      }
#pragma unroll
      for (int vv = 0; vv < 8; ++vv) {
        int pix = p0 + ms * 16 + vv + hi * 8;
        float r = fmaxf(acc[i][vv] * scale + shift, 0.f);
        featOut[(long)pix * 64 + n] = (_Float16)r;
      }
    }
  }
}

// ---------------------------------------------------------------------------
extern "C" void kernel_launch(void* const* d_in, const int* in_sizes, int n_in,
                              void* d_out, int out_size, void* d_ws, size_t ws_size,
                              hipStream_t stream) {
  const float* x         = (const float*)d_in[0];
  const float* off_w_cls = (const float*)d_in[1];
  const float* off_b_cls = (const float*)d_in[2];
  const float* dcn_w_cls = (const float*)d_in[3];
  const float* off_w_reg = (const float*)d_in[4];
  const float* off_b_reg = (const float*)d_in[5];
  const float* dcn_w_reg = (const float*)d_in[6];
  const float* cls_w1    = (const float*)d_in[7];
  const float* cls_b1    = (const float*)d_in[8];
  const float* bn_g      = (const float*)d_in[9];
  const float* bn_b      = (const float*)d_in[10];
  const float* bn_m      = (const float*)d_in[11];
  const float* bn_v      = (const float*)d_in[12];
  const float* cls_w2    = (const float*)d_in[13];
  const float* cls_b2    = (const float*)d_in[14];
  float* out = (float*)d_out;

  // workspace carve (256B aligned)
  char* ws = (char*)d_ws;
  size_t o = 0;
  auto carve = [&](size_t bytes) { char* p = ws + o; o += (bytes + 255) & ~(size_t)255; return p; };
  float*     off_buf  = (float*)    carve((size_t)NPIX * 72 * 4);   // 29.5 MB
  _Float16*  cfeat    = (_Float16*) carve((size_t)NPIX * 64 * 2);   // 13.1 MB
  _Float16*  rfeat    = (_Float16*) carve((size_t)NPIX * 64 * 2);
  _Float16*  tfeat    = (_Float16*) carve((size_t)NPIX * 64 * 2);   // cls hidden / head hidden
  _Float16*  packW    = (_Float16*) carve((size_t)18 * 4 * 512 * 2); // 72 KB

  const int GB = NPIX / 64;                 // 1600 GEMM blocks (64 px each)
  const long offN = (long)NPIX * 72;
  const int offBlocks = (int)((offN + 255) / 256);
  auto packBlocks = [](int K, int NT) { return ((K >> 5) * NT * 512 + 255) / 256; };

  // ---- cls deform branch ----
  pack_wB<<<packBlocks(576, 4), 256, 0, stream>>>(dcn_w_cls, packW, 64, 576, 4);
  offset_conv<<<offBlocks, 256, 0, stream>>>(x, off_w_cls, off_b_cls, off_buf);
  wmma_gemm<<<GB, 128, 0, stream>>>(0, x, off_buf, nullptr, packW, 576, 4,
      /*epi*/0, cfeat, nullptr, 0, 0, nullptr, nullptr, nullptr, nullptr, nullptr);

  // ---- reg deform branch ----
  pack_wB<<<packBlocks(576, 4), 256, 0, stream>>>(dcn_w_reg, packW, 64, 576, 4);
  offset_conv<<<offBlocks, 256, 0, stream>>>(x, off_w_reg, off_b_reg, off_buf);
  wmma_gemm<<<GB, 128, 0, stream>>>(0, x, off_buf, nullptr, packW, 576, 4,
      0, rfeat, nullptr, 0, 0, nullptr, nullptr, nullptr, nullptr, nullptr);

  // ---- cls tower: conv3x3 + BN + ReLU, then conv3x3 -> out[0:3] ----
  pack_wB<<<packBlocks(576, 4), 256, 0, stream>>>(cls_w1, packW, 64, 576, 4);
  wmma_gemm<<<GB, 128, 0, stream>>>(1, nullptr, nullptr, cfeat, packW, 576, 4,
      /*epi*/1, tfeat, nullptr, 0, 0, cls_b1, bn_g, bn_b, bn_m, bn_v);
  pack_wB<<<packBlocks(576, 1), 256, 0, stream>>>(cls_w2, packW, 3, 576, 1);
  wmma_gemm<<<GB, 128, 0, stream>>>(1, nullptr, nullptr, tfeat, packW, 576, 1,
      /*epi*/2, nullptr, out, /*chOff*/0, /*nReal*/3, cls_b2, nullptr, nullptr, nullptr, nullptr);

  // ---- heads on reg_feat: 1x1 (64->64,+b,ReLU) then 1x1 (64->c,+b) ----
  const int headC[5] = {2, 1, 3, 2, 2};
  int chOff = 3;
  for (int hI = 0; hI < 5; ++hI) {
    const float* w1 = (const float*)d_in[15 + 4 * hI + 0];
    const float* b1 = (const float*)d_in[15 + 4 * hI + 1];
    const float* w2 = (const float*)d_in[15 + 4 * hI + 2];
    const float* b2 = (const float*)d_in[15 + 4 * hI + 3];
    pack_wB<<<packBlocks(64, 4), 256, 0, stream>>>(w1, packW, 64, 64, 4);
    wmma_gemm<<<GB, 128, 0, stream>>>(2, nullptr, nullptr, rfeat, packW, 64, 4,
        0, tfeat, nullptr, 0, 0, b1, nullptr, nullptr, nullptr, nullptr);
    pack_wB<<<packBlocks(64, 1), 256, 0, stream>>>(w2, packW, headC[hI], 64, 1);
    wmma_gemm<<<GB, 128, 0, stream>>>(2, nullptr, nullptr, tfeat, packW, 64, 1,
        2, nullptr, out, chOff, headC[hI], b2, nullptr, nullptr, nullptr, nullptr);
    chOff += headC[hI];
  }
}
